// SDFNet_65506841199156
// MI455X (gfx1250) — matrix operand order repaired and checked by reading, not compiled, not run on hardware
//
#include <hip/hip_runtime.h>
#include <hip/hip_bf16.h>

typedef _Float16 half8_t __attribute__((ext_vector_type(8)));
typedef _Float16 v16h    __attribute__((ext_vector_type(16)));
typedef float    v8f     __attribute__((ext_vector_type(8)));
typedef unsigned int u32x4 __attribute__((ext_vector_type(4)));
typedef int      i32x8   __attribute__((ext_vector_type(8)));
typedef int      i32x4   __attribute__((ext_vector_type(4)));

#define LATENT 128
#define FDIM   32
#define GDIM   513
#define HID    256
#define XDIM   131     // LATENT + 3
#define KPAD   256     // padded SDF_IN (227 -> 256)
#define MT_PTS 64      // points per block

// ---------------------------------------------------------------------------
// Prep: W0 [227,256] -> w0t f16 col-major [n][k] zero-padded to 256 K,
//       W1 [256,256] -> w1t f16 col-major [n][k].  131072 elements total.
// ---------------------------------------------------------------------------
__global__ void prep_weights(const float* __restrict__ W0,
                             const float* __restrict__ W1,
                             _Float16* __restrict__ wsh) {
    int i = blockIdx.x * blockDim.x + threadIdx.x;
    if (i >= 2 * HID * KPAD) return;
    int mat = i >> 16;          // 0 = W0t, 1 = W1t
    int idx = i & 0xFFFF;
    int nn  = idx >> 8;         // output neuron (column of W)
    int kk  = idx & 255;        // input dim
    float v;
    if (mat == 0) v = (kk < 227) ? W0[kk * HID + nn] : 0.0f;
    else          v = W1[kk * HID + nn];
    wsh[i] = (_Float16)v;
}

// ---------------------------------------------------------------------------
// TDM: DMA a 256x256 f16 tile (row-major contiguous) from global into LDS.
// D# per CDNA5 ISA 8.3/8.4: group0 {count=1, lds_addr, global_addr, type=2},
// group1 {data_size=2B, tensor_dim0/1=256, tile_dim0/1=256, dim0_stride=256}.
// 6-arg builtin form (clang-23 / therock-10.0 headers).
// ---------------------------------------------------------------------------
__device__ __forceinline__ void tdm_load_256x256_f16(void* ldsDst, const void* gsrc) {
    unsigned long long ga = (unsigned long long)(uintptr_t)gsrc;
    unsigned int ldsOff   = (unsigned int)(uintptr_t)ldsDst;  // low 32 bits of flat = LDS offset
    u32x4 g0;
    g0[0] = 1u;                                   // count=1, user mode
    g0[1] = ldsOff;                               // lds_addr
    g0[2] = (unsigned int)ga;                     // global_addr[31:0]
    g0[3] = (unsigned int)((ga >> 32) & 0x1FFFFFFull) | 0x80000000u; // addr[56:32] | type=2
    i32x8 g1;
    g1[0] = 0x00010000;                           // workgroup_mask=0, data_size=1 (2 bytes)
    g1[1] = (int)(256u << 16);                    // tensor_dim0 lo16 = 256
    g1[2] = (int)(256u << 16);                    // dim0 hi=0 | tensor_dim1 lo16 = 256
    g1[3] = (int)(256u << 16);                    // dim1 hi=0 | tile_dim0 = 256
    g1[4] = 256;                                  // tile_dim1=256, tile_dim2=0
    g1[5] = 256;                                  // tensor_dim0_stride lo32 = 256
    g1[6] = 0;
    g1[7] = 0;
    i32x4 z4 = {0, 0, 0, 0};
    i32x8 z8 = {0, 0, 0, 0, 0, 0, 0, 0};
    __builtin_amdgcn_tensor_load_to_lds(g0, g1, z4, z4, z8, 0);
}

// ---------------------------------------------------------------------------
// One hidden layer: [64,256] f16 (LDS) x [256,256] f16 col-major (LDS)
// -> bias + relu -> [64,256] f16 (LDS).  Wave w: M-tile w>>1, N-tiles
// (w&1)*8 .. +7.  v_wmma_f32_16x16x32_f16, K-loop of 8.
// ---------------------------------------------------------------------------
__device__ __forceinline__ void gemm_relu_layer(const _Float16* __restrict__ Ain,
                                                const _Float16* __restrict__ Wcol,
                                                const float* __restrict__ bias,
                                                _Float16* __restrict__ Hout,
                                                int m0, int nb, int lrow, int lhi) {
    v8f acc[8] = {};
#pragma unroll
    for (int kk = 0; kk < 8; ++kk) {
        // A fragment (16-bit A layout: lanes<16 K 0..7 & 16..23; lanes>=16 K 8..15 & 24..31)
        const _Float16* ap = Ain + (m0 + lrow) * KPAD + kk * 32 + lhi * 8;
        half8_t alo = *(const half8_t*)ap;
        half8_t ahi = *(const half8_t*)(ap + 16);
        v16h af;
#pragma unroll
        for (int t = 0; t < 8; ++t) { af[t] = alo[t]; af[t + 8] = ahi[t]; }
#pragma unroll
        for (int j = 0; j < 8; ++j) {
            // B fragment: lane holds its N column, K 0..7 & 16..23 (lanes<16) / 8..15 & 24..31 (lanes>=16) (col-major W)
            const _Float16* bp = Wcol + ((nb + j) * 16 + lrow) * KPAD + kk * 32 + lhi * 8;
            half8_t blo = *(const half8_t*)bp;
            half8_t bhi = *(const half8_t*)(bp + 16);
            v16h bf;
#pragma unroll
            for (int t = 0; t < 8; ++t) { bf[t] = blo[t]; bf[t + 8] = bhi[t]; }
            acc[j] = __builtin_amdgcn_wmma_f32_16x16x32_f16(
                false, af, false, bf, (short)0, acc[j], false, false);
        }
    }
    // Writeback: C/D layout -> lanes<16 rows v (0..7), lanes>=16 rows 8+v; N = lane&15.
#pragma unroll
    for (int j = 0; j < 8; ++j) {
        int col = (nb + j) * 16 + lrow;
        float b = bias[col];
#pragma unroll
        for (int vv = 0; vv < 8; ++vv) {
            int row = m0 + lhi * 8 + vv;
            float val = acc[j][vv] + b;
            Hout[row * KPAD + col] = (_Float16)fmaxf(val, 0.0f);
        }
    }
}

// ---------------------------------------------------------------------------
// Fused SDF forward: triplane gather -> 2x GEMM+ReLU (WMMA) -> dot -> out
// ---------------------------------------------------------------------------
__global__ __launch_bounds__(256)
void sdf_forward(const float* __restrict__ x,
                 const float* __restrict__ P,
                 const float* __restrict__ b0,
                 const float* __restrict__ b1,
                 const float* __restrict__ W2,
                 const float* __restrict__ b2,
                 const _Float16* __restrict__ wsh,
                 float* __restrict__ out, int n) {
    extern __shared__ char smem[];
    _Float16* Abuf = (_Float16*)smem;            // 64 x 256 activations (f16)
    _Float16* Hbuf = Abuf + MT_PTS * KPAD;       // 64 x 256 hidden
    _Float16* Wbuf = Hbuf + MT_PTS * KPAD;       // 256 x 256 weights (col-major)

    const int tid  = threadIdx.x;
    const int wave = tid >> 5;
    const int lane = tid & 31;
    const int pt0  = blockIdx.x * MT_PTS;

    // Kick off W0t DMA into LDS while we build the activation tile.
    if (wave == 0) tdm_load_256x256_f16(Wbuf, wsh);

    // Zero the A tile (covers K padding 227..255 and out-of-range points).
    unsigned int* A32 = (unsigned int*)Abuf;
    for (int i = tid; i < MT_PTS * (KPAD / 2); i += 256) A32[i] = 0u;
    __syncthreads();

    // Latents + raw coords -> cols 0..130
    for (int i = tid; i < MT_PTS * XDIM; i += 256) {
        int r = i / XDIM, c = i - r * XDIM;
        int pt = pt0 + r;
        float v = (pt < n) ? x[(long)pt * XDIM + c] : 0.0f;
        Abuf[r * KPAD + c] = (_Float16)v;
    }

    // Tri-plane bilinear features -> cols 131..226 (P is L2-resident: 101MB < 192MB)
    for (int i = tid; i < MT_PTS * 96; i += 256) {
        int r = i & 63, f = i >> 6;
        int pt = pt0 + r;
        if (pt < n) {
            int pl = f >> 5, ch = f & 31;
            const float* xr = x + (long)pt * XDIM + LATENT;
            float c0 = xr[0], c1 = xr[1], c2 = xr[2];
            float u = (pl == 2) ? c1 : c0;
            float v = (pl == 0) ? c1 : c2;
            float gx = fminf(fmaxf((u + 1.0f) * 256.0f, 0.0f), 512.0f);
            float gy = fminf(fmaxf((v + 1.0f) * 256.0f, 0.0f), 512.0f);
            float x0f = floorf(gx), y0f = floorf(gy);
            int ix0 = (int)x0f, iy0 = (int)y0f;
            int ix1 = min(ix0 + 1, GDIM - 1), iy1 = min(iy0 + 1, GDIM - 1);
            float wx = gx - x0f, wy = gy - y0f;
            const float* pp = P + (long)(pl * FDIM + ch) * (GDIM * GDIM);
            float v00 = pp[iy0 * GDIM + ix0];
            float v01 = pp[iy0 * GDIM + ix1];
            float v10 = pp[iy1 * GDIM + ix0];
            float v11 = pp[iy1 * GDIM + ix1];
            float top = v00 + (v01 - v00) * wx;
            float bot = v10 + (v11 - v10) * wx;
            Abuf[r * KPAD + XDIM + f] = (_Float16)(top + (bot - top) * wy);
        }
    }

    if (wave == 0) __builtin_amdgcn_s_wait_tensorcnt(0);   // W0t landed in LDS
    __syncthreads();

    const int m0   = (wave >> 1) * 16;
    const int nb   = (wave & 1) * 8;
    const int lrow = lane & 15;
    const int lhi  = lane >> 4;

    // Layer 0: Abuf x W0t -> Hbuf
    gemm_relu_layer(Abuf, Wbuf, b0, Hbuf, m0, nb, lrow, lhi);
    __syncthreads();                                       // Wbuf reads done

    // Stage W1t, then Layer 1: Hbuf x W1t -> Abuf (reuse)
    if (wave == 0) {
        tdm_load_256x256_f16(Wbuf, wsh + HID * KPAD);
        __builtin_amdgcn_s_wait_tensorcnt(0);
    }
    __syncthreads();
    gemm_relu_layer(Hbuf, Wbuf, b1, Abuf, m0, nb, lrow, lhi);
    __syncthreads();

    // Final [256]->1: 4 lanes per point, shfl_xor reduction (wave32)
    {
        int r = tid >> 2, p = tid & 3;
        const _Float16* hp = Abuf + r * KPAD + p * 64;
        const float* w2p = W2 + p * 64;
        float s = 0.0f;
#pragma unroll
        for (int c = 0; c < 64; ++c) s += (float)hp[c] * w2p[c];
        s += __shfl_xor(s, 1, 32);
        s += __shfl_xor(s, 2, 32);
        int pt = pt0 + r;
        if (p == 0 && pt < n) out[pt] = s + b2[0];
    }
}

extern "C" void kernel_launch(void* const* d_in, const int* in_sizes, int n_in,
                              void* d_out, int out_size, void* d_ws, size_t ws_size,
                              hipStream_t stream) {
    const float* x  = (const float*)d_in[0];
    const float* P  = (const float*)d_in[1];
    const float* W0 = (const float*)d_in[2];
    const float* b0 = (const float*)d_in[3];
    const float* W1 = (const float*)d_in[4];
    const float* b1 = (const float*)d_in[5];
    const float* W2 = (const float*)d_in[6];
    const float* b2 = (const float*)d_in[7];
    float* out = (float*)d_out;
    _Float16* wsh = (_Float16*)d_ws;   // 131072 halves = 256KB (w0t | w1t)

    int n = in_sizes[0] / XDIM;        // 500000

    prep_weights<<<(2 * HID * KPAD + 255) / 256, 256, 0, stream>>>(W0, W1, wsh);

    int nblocks = (n + MT_PTS - 1) / MT_PTS;
    size_t shmem = (size_t)(MT_PTS * KPAD * 2 + HID * KPAD) * sizeof(_Float16); // 192 KB
    sdf_forward<<<nblocks, 256, shmem, stream>>>(x, P, b0, b1, W2, b2, wsh, out, n);
}
